// EDPConv_58909771432453
// MI455X (gfx1250) — matrix operand order repaired and checked by reading, not compiled, not run on hardware
//
#include <hip/hip_runtime.h>
#include <hip/hip_bf16.h>

typedef __attribute__((ext_vector_type(16))) _Float16 v16h;
typedef __attribute__((ext_vector_type(8)))  float    v8f;

#define N_NODES 1024
#define C_CH    4
#define E_EDGES 32768
#define F_IN    64
#define F_OUT   64
#define H_NODE  256
#define H_EDGE  128
#define C_OUT   4

// ---------- TDM availability (guarded so compile never breaks) ----------
#ifndef USE_TDM
#if defined(__has_builtin)
#if __has_builtin(__builtin_amdgcn_tensor_load_to_lds) && __has_builtin(__builtin_amdgcn_s_wait_tensorcnt)
#define USE_TDM 1
#endif
#endif
#endif
#ifndef USE_TDM
#define USE_TDM 0
#endif

#if USE_TDM
typedef __attribute__((ext_vector_type(4))) unsigned int u32x4;
typedef __attribute__((ext_vector_type(4))) int          i32x4;
typedef __attribute__((ext_vector_type(8))) int          i32x8;

// 1-D contiguous TDM copy: nfloats f32 from global -> LDS (CDNA5 ISA)
__device__ __forceinline__ void tdm_load_1d(const float* gsrc, unsigned lds_off, unsigned nfloats) {
  unsigned long long ga = (unsigned long long)(size_t)gsrc;
  u32x4 g0;
  g0[0] = 1u;                                              // count=1, user descriptor
  g0[1] = lds_off;                                         // lds_addr (bytes)
  g0[2] = (unsigned)ga;                                    // global_addr[31:0]
  g0[3] = (unsigned)((ga >> 32) & 0x01ffffffull) | 0x80000000u; // addr[56:32] | type=2
  i32x8 g1;
  g1[0] = 0x00020000;                                      // data_size=4B, no multicast
  g1[1] = (int)(nfloats << 16);                            // tensor_dim0[15:0]
  g1[2] = (int)(((nfloats >> 16) & 0xffffu) | (1u << 16)); // tensor_dim0[31:16], tensor_dim1=1
  g1[3] = (int)(nfloats << 16);                            // tile_dim0
  g1[4] = 0;                                               // tile_dim1/2 unused
  g1[5] = (int)nfloats;                                    // tensor_dim0_stride (lo)
  g1[6] = 0;
  g1[7] = 0;
  i32x4 z4 = {0, 0, 0, 0};
#if __clang_major__ >= 23
  i32x8 z8 = {0, 0, 0, 0, 0, 0, 0, 0};
  __builtin_amdgcn_tensor_load_to_lds(g0, g1, z4, z4, z8, 0);
#else
  __builtin_amdgcn_tensor_load_to_lds(g0, g1, z4, z4, 0);
#endif
}
#endif

// ---------- WMMA fragment layout helpers (CDNA5 ISA §7.12.2) ----------
__device__ __forceinline__ int a_koff(int e, int hi) {
  return ((e & 8) << 1) + (hi << 3) + (e & 7);             // 8-consecutive runs per half
}
__device__ __forceinline__ int aswz_off(int m, int k, int KC) {
  int rt = m >> 4, r = m & 15;
  int kc = k >> 5, ko = k & 31;
  int hi = (ko >> 3) & 1;
  int e  = ((ko >> 4) << 3) | (ko & 7);
  return (((rt * KC + kc) << 5) + (hi << 4) + r) * 16 + e;
}

// ---------- prep: pack B [K,N] (f32 row-major, N padded, realN valid) into frag layout ----------
__global__ void k_bswz(const float* __restrict__ B, int KC, int realN,
                       _Float16* __restrict__ out, int total) {
  int i = blockIdx.x * blockDim.x + threadIdx.x;
  if (i >= total) return;
  int e = i & 15, lane = (i >> 4) & 31, idx = i >> 9;
  int kc = idx % KC, tn = idx / KC;
  int k   = (kc << 5) + ((e & 8) << 1) + ((lane >> 4) << 3) + (e & 7);
  int col = (tn << 4) + (lane & 15);
  out[i] = (_Float16)((col < realN) ? B[k * realN + col] : 0.f);
}

// ---------- prep: pack A [M,K] f32 row-major into A-fragment layout ----------
__global__ void k_aswz(const float* __restrict__ A, int KC,
                       _Float16* __restrict__ out, int total) {
  int i = blockIdx.x * blockDim.x + threadIdx.x;
  if (i >= total) return;
  int e = i & 15, lane = (i >> 4) & 31, idx = i >> 9;
  int kc = idx % KC, rt = idx / KC;
  int r = lane & 15, hi = lane >> 4;
  int K = KC << 5;
  int k = (kc << 5) + a_koff(e, hi);
  out[i] = (_Float16)A[((rt << 4) + r) * K + k];
}

// ---------- K0a: zero ----------
__global__ void k_zero_f32(float* p, int n) {
  int i = blockIdx.x * blockDim.x + threadIdx.x;
  if (i < n) p[i] = 0.f;
}

// ---------- K0b: out_cat[n][c*64+f] = (1+eps)*x[n][f] ----------
__global__ void k_init_outcat(const float* __restrict__ x, const float* __restrict__ eps,
                              float* __restrict__ out_cat) {
  int i = blockIdx.x * blockDim.x + threadIdx.x;
  if (i >= N_NODES * C_CH * F_IN) return;
  int f  = i & (F_IN - 1);
  int nd = i >> 8;
  out_cat[i] = (1.f + eps[0]) * x[nd * F_IN + f];
}

// ---------- K1: weighted scatter-add + dense adjacency ----------
__global__ void k_edge_scatter(const float* __restrict__ x, const float* __restrict__ ew,
                               const int* __restrict__ ei,
                               float* __restrict__ out_cat, float* __restrict__ a_dense) {
  int grp = blockIdx.x * 4 + (threadIdx.x >> 6);
  int l   = threadIdx.x & 63;
  int c   = grp >> 15;                          // E = 2^15
  int e   = grp & (E_EDGES - 1);
  float w = ew[c * E_EDGES + e];
  int s   = ei[(c * 2 + 0) * E_EDGES + e];
  int d   = ei[(c * 2 + 1) * E_EDGES + e];
  atomicAdd(&out_cat[d * (C_CH * F_IN) + c * F_IN + l], x[s * F_IN + l] * w);
  if (l == 0) atomicAdd(&a_dense[(s * N_NODES + d) * C_CH + c], w);
}

// ---------- K2: H = relu(out_cat @ W1n + b1n) -> A-swizzled f16 ----------
__global__ void k_node_mlp1(const _Float16* __restrict__ Asw, const _Float16* __restrict__ Bsw,
                            const float* __restrict__ bias, _Float16* __restrict__ Hsw) {
  int wave = threadIdx.x >> 5;
  int tile = blockIdx.x * 4 + wave;            // 64 x 16 tiles
  int rt = tile >> 4, ct = tile & 15;
  int lane = threadIdx.x & 31;
  int hi = lane >> 4, n = lane & 15;
  v8f acc = {};
#pragma unroll
  for (int ks = 0; ks < 8; ++ks) {
    v16h af = *(const v16h*)(Asw + (((rt * 8 + ks) << 5) + lane) * 16);
    v16h bf = *(const v16h*)(Bsw + (((ct * 8 + ks) << 5) + lane) * 16);
    acc = __builtin_amdgcn_wmma_f32_16x16x32_f16(false, af, false, bf, (short)0, acc, false, false);
  }
  float bv = bias[(ct << 4) + n];
#pragma unroll
  for (int g = 0; g < 8; ++g) {
    int m = (rt << 4) + (hi << 3) + g;
    int k = (ct << 4) + n;                     // output col == next layer's K
    float v = acc[g] + bv;
    v = v > 0.f ? v : 0.f;
    Hsw[aswz_off(m, k, 8)] = (_Float16)v;
  }
}

// ---------- K3: x_next = H @ W2n + b2n -> f32 (d_out) + A-swizzled f16 ----------
__global__ void k_node_mlp2(const _Float16* __restrict__ Hsw, const _Float16* __restrict__ Bsw,
                            const float* __restrict__ bias,
                            float* __restrict__ xn, _Float16* __restrict__ xnh_sw) {
  int wave = threadIdx.x >> 5;
  int tile = blockIdx.x * 4 + wave;            // 64 x 4 tiles
  int rt = tile >> 2, ct = tile & 3;
  int lane = threadIdx.x & 31;
  int hi = lane >> 4, n = lane & 15;
  v8f acc = {};
#pragma unroll
  for (int ks = 0; ks < 8; ++ks) {
    v16h af = *(const v16h*)(Hsw + (((rt * 8 + ks) << 5) + lane) * 16);
    v16h bf = *(const v16h*)(Bsw + (((ct * 8 + ks) << 5) + lane) * 16);
    acc = __builtin_amdgcn_wmma_f32_16x16x32_f16(false, af, false, bf, (short)0, acc, false, false);
  }
  float bv = bias[(ct << 4) + n];
#pragma unroll
  for (int g = 0; g < 8; ++g) {
    int m = (rt << 4) + (hi << 3) + g;
    int col = (ct << 4) + n;
    float v = acc[g] + bv;
    xn[m * F_OUT + col] = v;
    xnh_sw[aswz_off(m, col, 2)] = (_Float16)v;
  }
}

// ---------- K4: U = x@W1e[4:68];  V = x@W1e[68:132] + b1e (bias folded in) ----------
__global__ void k_uv_gemm(const _Float16* __restrict__ Xsw,
                          const _Float16* __restrict__ BUsw, const _Float16* __restrict__ BVsw,
                          const float* __restrict__ b1e,
                          float* __restrict__ U, float* __restrict__ V) {
  int wave = threadIdx.x >> 5;
  int tile = blockIdx.x * 4 + wave;            // 2 * (64 x 8) tiles
  int isV = tile >= 512;
  int t = tile & 511;
  int rt = t >> 3, ct = t & 7;
  const _Float16* Bsw = isV ? BVsw : BUsw;
  float* O = isV ? V : U;
  int lane = threadIdx.x & 31;
  int hi = lane >> 4, n = lane & 15;
  v8f acc = {};
#pragma unroll
  for (int ks = 0; ks < 2; ++ks) {
    v16h af = *(const v16h*)(Xsw + (((rt * 2 + ks) << 5) + lane) * 16);
    v16h bf = *(const v16h*)(Bsw + (((ct * 2 + ks) << 5) + lane) * 16);
    acc = __builtin_amdgcn_wmma_f32_16x16x32_f16(false, af, false, bf, (short)0, acc, false, false);
  }
  float bv = isV ? b1e[(ct << 4) + n] : 0.f;
#pragma unroll
  for (int g = 0; g < 8; ++g)
    O[((rt << 4) + (hi << 3) + g) * H_EDGE + (ct << 4) + n] = acc[g] + bv;
}

// ---------- K5: fused edge MLP + symmetrize.
// Block (bi<=bj): waves 0-3 compute tile T1=(i,j) in regs; waves 4-7 compute mirror
// T2=(j,i) into LDS; combine T1[ii,jj]+T2[jj,ii]+2*b2e and write both orientations. ----------
__global__ void k_edge_mlp(const float* __restrict__ U, const float* __restrict__ Vb,
                           const float* __restrict__ a_dense,
                           const float* __restrict__ W1e,
                           const _Float16* __restrict__ W2sw, const float* __restrict__ b2e,
                           float* __restrict__ adj) {
  __shared__ float sUi[16 * H_EDGE], sVj[16 * H_EDGE];
  __shared__ float sUj[16 * H_EDGE], sVi[16 * H_EDGE];
  __shared__ float sC[4 * H_EDGE];             // W1e rows 0..3
  __shared__ float sT2[16 * 16 * C_OUT];       // mirror-tile results [q][p][ch]
  int bi = blockIdx.y, bj = blockIdx.x;
  if (bi > bj) return;                         // uniform early exit (before any barrier)
  int i0 = bi << 4, j0 = bj << 4;
  int lane = threadIdx.x & 31;
  int wave = threadIdx.x >> 5;
  int r = lane & 15, hi = lane >> 4, n = r;

  for (int t = threadIdx.x; t < 4 * H_EDGE; t += 256) sC[t] = W1e[t];

#if USE_TDM
  if (wave == 0) {
    tdm_load_1d(U  + i0 * H_EDGE, (unsigned)(size_t)(void*)sUi, 16 * H_EDGE);
    tdm_load_1d(Vb + j0 * H_EDGE, (unsigned)(size_t)(void*)sVj, 16 * H_EDGE);
    tdm_load_1d(U  + j0 * H_EDGE, (unsigned)(size_t)(void*)sUj, 16 * H_EDGE);
    tdm_load_1d(Vb + i0 * H_EDGE, (unsigned)(size_t)(void*)sVi, 16 * H_EDGE);
    __builtin_amdgcn_s_wait_tensorcnt(0);
  }
#else
  for (int t = threadIdx.x; t < (16 * H_EDGE) / 4; t += 256) {
    ((float4*)sUi)[t] = ((const float4*)(U  + i0 * H_EDGE))[t];
    ((float4*)sVj)[t] = ((const float4*)(Vb + j0 * H_EDGE))[t];
    ((float4*)sUj)[t] = ((const float4*)(U  + j0 * H_EDGE))[t];
    ((float4*)sVi)[t] = ((const float4*)(Vb + i0 * H_EDGE))[t];
  }
#endif
  __syncthreads();

  v16h bfr[4];                                 // W2e padded [128,16] B-fragments
#pragma unroll
  for (int ks = 0; ks < 4; ++ks)
    bfr[ks] = *(const v16h*)(W2sw + ((ks << 5) + lane) * 16);
  float b2v = (n < C_OUT) ? b2e[n] : 0.f;

  int isT2 = wave >> 2;                        // waves 4-7 compute mirror tile
  const float* su = isT2 ? sUj : sUi;
  const float* sv = isT2 ? sVi : sVj;
  int rowbase = isT2 ? j0 : i0;
  int colbase = isT2 ? i0 : j0;
  int w4 = (wave & 3) << 2;

  v8f acc[4];
#pragma unroll
  for (int cc = 0; cc < 4; ++cc) {
    int ii = w4 + cc;                          // row within owning block
    const float* u = su + ii * H_EDGE;
    const float* v = sv + r * H_EDGE;          // A-frag row r == col within tile
    float4 av = *(const float4*)(a_dense + (((rowbase + ii) << 10) + colbase + r) * C_CH);
    acc[cc] = (v8f){};
#pragma unroll
    for (int ks = 0; ks < 4; ++ks) {
      v16h af;
      int b0 = (ks << 5) + (hi << 3);
#pragma unroll
      for (int half = 0; half < 2; ++half) {
        int k0 = b0 + half * 16;               // 8-consecutive k per half -> b128 LDS loads
        int e0 = half * 8;
#pragma unroll
        for (int q = 0; q < 8; ++q) {
          int k = k0 + q;
          float h = u[k] + v[k]                 // b1e folded into Vb
                  + av.x * sC[k]              + av.y * sC[H_EDGE + k]
                  + av.z * sC[2 * H_EDGE + k] + av.w * sC[3 * H_EDGE + k];
          af[e0 + q] = (_Float16)(h > 0.f ? h : 0.f);
        }
      }
      acc[cc] = __builtin_amdgcn_wmma_f32_16x16x32_f16(false, af, false, bfr[ks], (short)0, acc[cc], false, false);
    }
  }

  // mirror tile -> LDS: sT2[(q*16 + p)*4 + ch], q = row in j-block, p = col in i-block
  if (isT2 && n < C_OUT) {
#pragma unroll
    for (int cc = 0; cc < 4; ++cc) {
      int q = w4 + cc;
#pragma unroll
      for (int g = 0; g < 8; ++g) {
        int p = (hi << 3) + g;
        sT2[((q << 4) + p) * C_OUT + n] = acc[cc][g];
      }
    }
  }
  __syncthreads();

  // combine + write both orientations
  if (!isT2 && n < C_OUT) {
#pragma unroll
    for (int cc = 0; cc < 4; ++cc) {
      int ii = w4 + cc;
#pragma unroll
      for (int g = 0; g < 8; ++g) {
        int jj = (hi << 3) + g;
        float val = acc[cc][g] + sT2[((jj << 4) + ii) * C_OUT + n] + 2.f * b2v;
        adj[(((i0 + ii) << 10) + j0 + jj) * C_OUT + n] = val;
        adj[(((j0 + jj) << 10) + i0 + ii) * C_OUT + n] = val;
      }
    }
  }
}

extern "C" void kernel_launch(void* const* d_in, const int* in_sizes, int n_in,
                              void* d_out, int out_size, void* d_ws, size_t ws_size,
                              hipStream_t stream) {
  (void)in_sizes; (void)n_in; (void)out_size; (void)ws_size;
  const float* x   = (const float*)d_in[0];
  const float* ew  = (const float*)d_in[1];
  const float* W1n = (const float*)d_in[2];
  const float* b1n = (const float*)d_in[3];
  const float* W2n = (const float*)d_in[4];
  const float* b2n = (const float*)d_in[5];
  const float* W1e = (const float*)d_in[6];
  const float* b1e = (const float*)d_in[7];
  const float* W2e = (const float*)d_in[8];
  const float* b2e = (const float*)d_in[9];
  const float* eps = (const float*)d_in[10];
  const int*   ei  = (const int*)d_in[11];

  char* ws = (char*)d_ws;                                       // ~19.4 MB used
  float*    a_dense = (float*)   (ws);                          // 16 MB [N,N,4]
  float*    out_cat = (float*)   (ws + (16u << 20));            // 1 MB
  float*    U       = (float*)   (ws + (17u << 20));            // 512 KB
  float*    V       = (float*)   (ws + (17u << 20) + (512u << 10));
  _Float16* Acat_sw = (_Float16*)(ws + (18u << 20));            // 512 KB
  _Float16* H_sw    = (_Float16*)(ws + (18u << 20) + (512u << 10));
  _Float16* xnh_sw  = (_Float16*)(ws + (19u << 20));            // 128 KB
  _Float16* W1n_sw  = (_Float16*)(ws + (19u << 20) + (128u << 10));
  _Float16* W2n_sw  = (_Float16*)(ws + (19u << 20) + (256u << 10));
  _Float16* W1eU_sw = (_Float16*)(ws + (19u << 20) + (288u << 10));
  _Float16* W1eV_sw = (_Float16*)(ws + (19u << 20) + (304u << 10));
  _Float16* W2e_sw  = (_Float16*)(ws + (19u << 20) + (320u << 10));

  float* xn      = (float*)d_out;                               // [1024,64]
  float* adj_out = (float*)d_out + N_NODES * F_OUT;             // [1024,1024,4]

  k_zero_f32    <<<(N_NODES * N_NODES * C_OUT + 255) / 256, 256, 0, stream>>>(a_dense, N_NODES * N_NODES * C_OUT);
  k_init_outcat <<<(N_NODES * C_CH * F_IN + 255) / 256, 256, 0, stream>>>(x, eps, out_cat);
  // weight pre-swizzle (fragment layout, f16)
  k_bswz<<<256, 256, 0, stream>>>(W1n, 8, H_NODE, W1n_sw, H_NODE * H_NODE);
  k_bswz<<< 64, 256, 0, stream>>>(W2n, 8, F_OUT, W2n_sw, H_NODE * F_OUT);
  k_bswz<<< 32, 256, 0, stream>>>(W1e +  4 * H_EDGE, 2, H_EDGE, W1eU_sw, F_OUT * H_EDGE);
  k_bswz<<< 32, 256, 0, stream>>>(W1e + 68 * H_EDGE, 2, H_EDGE, W1eV_sw, F_OUT * H_EDGE);
  k_bswz<<<  8, 256, 0, stream>>>(W2e, 4, C_OUT, W2e_sw, H_EDGE * 16);

  k_edge_scatter<<<(C_CH * E_EDGES) / 4, 256, 0, stream>>>(x, ew, ei, out_cat, a_dense);
  k_aswz        <<<1024, 256, 0, stream>>>(out_cat, 8, Acat_sw, N_NODES * H_NODE);

  k_node_mlp1<<<256, 128, 0, stream>>>(Acat_sw, W1n_sw, b1n, H_sw);
  k_node_mlp2<<< 64, 128, 0, stream>>>(H_sw, W2n_sw, b2n, xn, xnh_sw);
  k_uv_gemm  <<<256, 128, 0, stream>>>(xnh_sw, W1eU_sw, W1eV_sw, b1e, U, V);
  k_edge_mlp <<<dim3(N_NODES / 16, N_NODES / 16), 256, 0, stream>>>(U, V, a_dense, W1e, W2e_sw, b2e, adj_out);
}
